// SupervisedContrastiveLoss_31799937860140
// MI455X (gfx1250) — compile-verified, matrix operand order checked by reading, not run on hardware
//
#include <hip/hip_runtime.h>
#include <hip/hip_bf16.h>
#include <math.h>

typedef __attribute__((ext_vector_type(16))) _Float16 v16h;
typedef __attribute__((ext_vector_type(8)))  float    v8f;

#define BDIM   8192
#define DDIM   256
#define KSTEPS (DDIM / 32)   // 8 WMMA k-steps of 32
#define INV_T  10.0f         // 1 / temperature
#define EPS    1e-8f

// ---------------------------------------------------------------------------
// Fragment loader for V_WMMA_F32_16X16X32_F16 (ISA 7.12.2, 16-bit A 16x32):
//   lane in [0,15]  : row M=lane,  halves[0..7]=K[kb..kb+7],    halves[8..15]=K[kb+16..kb+23]
//   lane in [16,31] : row M=l-16,  halves[0..7]=K[kb+8..kb+15], halves[8..15]=K[kb+24..kb+31]
// B (32x16, column j of sim == row j of n16) mirrors this with lane -> N, so
// the same loader serves A and B for the n @ n^T product.
// ---------------------------------------------------------------------------
__device__ __forceinline__ v16h load_frag16(const _Float16* rowPtr, int kBase, int hi) {
    union { v16h v; float4 q[2]; } u;
    const float4* p = (const float4*)(rowPtr + kBase + (hi << 3)); // 16B-aligned
    u.q[0] = p[0];   // 8 halves: kBase + 8*hi + [0..7]
    u.q[1] = p[2];   // 8 halves: kBase + 16 + 8*hi + [0..7]
    return u.v;
}

__device__ __forceinline__ float reduce16(float v) {
    // sum across the 16 lanes of each wave half (bit4 of lane untouched)
    v += __shfl_xor(v, 1, 32);
    v += __shfl_xor(v, 2, 32);
    v += __shfl_xor(v, 4, 32);
    v += __shfl_xor(v, 8, 32);
    return v;
}

// ---------------------------------------------------------------------------
// K1: row L2-norm, write f16 normalized latents, zero per-row accumulators.
// One block (256 threads) per row.
// ---------------------------------------------------------------------------
__global__ void scl_normalize_kernel(const float* __restrict__ latents,
                                     _Float16* __restrict__ n16,
                                     float* __restrict__ denom,
                                     float* __restrict__ psum,
                                     float* __restrict__ pcnt) {
    const int row = blockIdx.x;
    const int t   = threadIdx.x;
    const float x = latents[(size_t)row * DDIM + t];

    float s = x * x;
    s += __shfl_xor(s, 1, 32);
    s += __shfl_xor(s, 2, 32);
    s += __shfl_xor(s, 4, 32);
    s += __shfl_xor(s, 8, 32);
    s += __shfl_xor(s, 16, 32);

    __shared__ float ws[8];
    __shared__ float totS;
    const int wave = t >> 5, lane = t & 31;
    if (lane == 0) ws[wave] = s;
    __syncthreads();
    if (t == 0) {
        float tt = 0.f;
        #pragma unroll
        for (int w = 0; w < 8; ++w) tt += ws[w];
        totS = tt;
    }
    __syncthreads();

    const float inv = 1.0f / fmaxf(sqrtf(totS), EPS);
    n16[(size_t)row * DDIM + t] = (_Float16)(x * inv);

    if (t == 0) { denom[row] = 0.f; psum[row] = 0.f; pcnt[row] = 0.f; }
}

// ---------------------------------------------------------------------------
// K2: fused n @ n^T GEMM (WMMA f16->f32) + exp/mask row reductions.
// Grid: BDIM/32 workgroups; each owns a 32-row strip across all 8192 columns.
// 8 waves = 2 (M) x 4 (N); wave tile 16x16 per column iteration (N-step 64).
// Column tiles are register double-buffered: while the 8 WMMAs + exp epilogue
// for tile jt run out of one fragment buffer, the 16 global_load_b128 for
// tile jt+64 are in flight into the other.
// ---------------------------------------------------------------------------
__global__ void __launch_bounds__(256)
scl_sim_kernel(const _Float16* __restrict__ n16,
               const int* __restrict__ labels,
               float* __restrict__ denom,
               float* __restrict__ psum,
               float* __restrict__ pcnt) {
    __shared__ int ldsLab[BDIM];           // 32 KB of the 320 KB WGP LDS

    const int t     = threadIdx.x;
    const int wave  = t >> 5;
    const int lane  = t & 31;
    const int wm    = wave >> 2;           // 0..1 : which 16-row half of strip
    const int wn    = wave & 3;            // 0..3 : which 16-col slice of N-tile
    const int nLane = lane & 15;
    const int hi    = lane >> 4;

    // stage labels into LDS once (L2-resident source)
    for (int i = t; i < BDIM; i += 256) ldsLab[i] = labels[i];
    __syncthreads();

    const int rowTile = blockIdx.x * 32 + wm * 16;

    // Preload all A fragments for this wave's 16 rows (K=256 -> 8 frags, 64 VGPRs)
    const _Float16* rowPtrA = n16 + (size_t)(rowTile + nLane) * DDIM;
    v16h aF[KSTEPS];
    #pragma unroll
    for (int kk = 0; kk < KSTEPS; ++kk) aF[kk] = load_frag16(rowPtrA, kk * 32, hi);

    // Row labels for the C-fragment rows this lane touches: M = r + 8*hi
    int labRow[8];
    #pragma unroll
    for (int r = 0; r < 8; ++r) labRow[r] = ldsLab[rowTile + hi * 8 + r];

    float dAcc[8], pAcc[8], cAcc[8];
    #pragma unroll
    for (int r = 0; r < 8; ++r) { dAcc[r] = 0.f; pAcc[r] = 0.f; cAcc[r] = 0.f; }

    // B-tile base pointer for this lane's column (wraps for the tail prefetch)
    auto bptr = [&](int jt) -> const _Float16* {
        return n16 + (size_t)((jt & (BDIM - 1)) + wn * 16 + nLane) * DDIM;
    };

    // 8 chained WMMAs + exp/mask epilogue for one 16x16 tile
    auto processTile = [&](int jt, const v16h* bF) {
        const int jcol = jt + wn * 16 + nLane;          // this lane's sim column
        v8f c = {};
        #pragma unroll
        for (int kk = 0; kk < KSTEPS; ++kk) {
            c = __builtin_amdgcn_wmma_f32_16x16x32_f16(
                    /*neg_a=*/false, aF[kk], /*neg_b=*/false, bF[kk],
                    /*c_mod=*/(short)0, c, /*reuse_a=*/false, /*reuse_b=*/false);
        }
        const int labCol = ldsLab[jcol];
        #pragma unroll
        for (int r = 0; r < 8; ++r) {
            const float sim  = c[r] * INV_T;            // cosine / temperature
            const int   i    = rowTile + hi * 8 + r;    // C layout: M = r + 8*hi
            const bool  self = (i == jcol);
            dAcc[r] += self ? 0.f : __expf(sim);        // denom excludes diagonal
            const bool pos = (!self) && (labRow[r] == labCol);
            pAcc[r] += pos ? sim : 0.f;
            cAcc[r] += pos ? 1.f : 0.f;
        }
    };

    // Register double-buffered column sweep (unrolled by 2 tiles)
    v16h bA[KSTEPS], bB[KSTEPS];
    {
        const _Float16* p0 = bptr(0);
        #pragma unroll
        for (int kk = 0; kk < KSTEPS; ++kk) bA[kk] = load_frag16(p0, kk * 32, hi);
    }
    for (int jt = 0; jt < BDIM; jt += 128) {
        const _Float16* pN = bptr(jt + 64);             // prefetch tile jt+64
        #pragma unroll
        for (int kk = 0; kk < KSTEPS; ++kk) bB[kk] = load_frag16(pN, kk * 32, hi);
        processTile(jt, bA);

        const _Float16* pN2 = bptr(jt + 128);           // prefetch tile jt+128 (wraps at end)
        #pragma unroll
        for (int kk = 0; kk < KSTEPS; ++kk) bA[kk] = load_frag16(pN2, kk * 32, hi);
        processTile(jt + 64, bB);
    }

    // Reduce each row across the 16 lanes holding its N slice, then the 4
    // wn-waves combine via a few global f32 atomics.
    #pragma unroll
    for (int r = 0; r < 8; ++r) {
        const float d  = reduce16(dAcc[r]);
        const float p  = reduce16(pAcc[r]);
        const float cc = reduce16(cAcc[r]);
        if (nLane == 0) {
            const int i = rowTile + hi * 8 + r;
            atomicAdd(&denom[i], d);
            atomicAdd(&psum[i],  p);
            atomicAdd(&pcnt[i],  cc);
        }
    }
}

// ---------------------------------------------------------------------------
// K3: per-row loss, masked mean -> scalar. Single block.
// ---------------------------------------------------------------------------
__global__ void scl_finalize_kernel(const float* __restrict__ denom,
                                    const float* __restrict__ psum,
                                    const float* __restrict__ pcnt,
                                    float* __restrict__ out) {
    const int t = threadIdx.x;
    float lsum = 0.f, lcnt = 0.f;
    for (int i = t; i < BDIM; i += 256) {
        const float c = pcnt[i];
        if (c > 0.f) {
            lsum += __logf(fmaxf(denom[i], EPS)) - psum[i] / c;
            lcnt += 1.f;
        }
    }
    #pragma unroll
    for (int m = 16; m > 0; m >>= 1) {
        lsum += __shfl_xor(lsum, m, 32);
        lcnt += __shfl_xor(lcnt, m, 32);
    }
    __shared__ float sS[8], sC[8];
    const int wave = t >> 5, lane = t & 31;
    if (lane == 0) { sS[wave] = lsum; sC[wave] = lcnt; }
    __syncthreads();
    if (t == 0) {
        float ts = 0.f, tc = 0.f;
        #pragma unroll
        for (int w = 0; w < 8; ++w) { ts += sS[w]; tc += sC[w]; }
        out[0] = ts / fmaxf(tc, 1.f);
    }
}

// ---------------------------------------------------------------------------
extern "C" void kernel_launch(void* const* d_in, const int* in_sizes, int n_in,
                              void* d_out, int out_size, void* d_ws, size_t ws_size,
                              hipStream_t stream) {
    (void)n_in; (void)out_size; (void)ws_size; (void)in_sizes;

    const float* latents = (const float*)d_in[0];
    const int*   labels  = (const int*)d_in[1];

    char* ws = (char*)d_ws;
    _Float16* n16 = (_Float16*)ws;
    const size_t n16Bytes = (size_t)BDIM * DDIM * sizeof(_Float16);
    float* denom = (float*)(ws + n16Bytes);
    float* psum  = denom + BDIM;
    float* pcnt  = psum + BDIM;

    scl_normalize_kernel<<<BDIM, DDIM, 0, stream>>>(latents, n16, denom, psum, pcnt);
    scl_sim_kernel<<<BDIM / 32, 256, 0, stream>>>(n16, labels, denom, psum, pcnt);
    scl_finalize_kernel<<<1, 256, 0, stream>>>(denom, psum, pcnt, (float*)d_out);
}
